// CrossFrameAttention_86397562126435
// MI455X (gfx1250) — compile-verified, matrix operand order checked by reading, not run on hardware
//
#include <hip/hip_runtime.h>
#include <hip/hip_bf16.h>
#include <stdint.h>

typedef __bf16 bf16_t;
typedef __bf16 v16bf __attribute__((ext_vector_type(16)));
typedef float  v8f   __attribute__((ext_vector_type(8)));
typedef unsigned int v4u __attribute__((ext_vector_type(4)));
typedef int v8i_t __attribute__((ext_vector_type(8)));
typedef int v4i_t __attribute__((ext_vector_type(4)));

#define C_DIM 128
#define NT    64          // tokens per block
#define HWSZ  (128 * 128) // H*W

// ---- LDS layout (bytes) ----
// sW    [0,      32768)  one 128x128 weight matrix, bf16, A-fragment swizzled
// sX    [32768,  81920)  3 frames x 128x64 bf16, B-fragment swizzled
// sXt   [81920, 114688)  f_t tile fp32 (residual), [c][t]  <- filled by TDM
// sQ    [114688,131072)  64x128 bf16, token-major
// sK    [131072,180224)  3 x 64x128 bf16
// sV    [180224,229376)  3 x 64x128 bf16   (recycled as fp32 Xln[128][64])
// sCtx  [229376,245760)  128x64 bf16, B-fragment swizzled
// sMu   [245760,246016)  64 f32
// sRs   [246016,246272)  64 f32
// sBias [246272,249344)  768 f32: in_b[384] | out_b[128] | gamma[128] | beta[128]
#define SMEM_BYTES 249344

// WMMA 16-bit fragment lane mapping (CDNA5 ISA 7.12.2):
// within a 32-wide K chunk, element r: group g = (r>>3)&1, half j = (r&7)|((r>>4)<<3)
__device__ __forceinline__ int swzW(int m, int c) {  // A fragment: W[m][c]
  int r = c & 31;
  int g = (r >> 3) & 1;
  int j = (r & 7) | ((r >> 4) << 3);
  int L = (g << 4) | (m & 15);
  return (((m >> 4) << 2) + (c >> 5)) * 512 + L * 16 + j;
}
__device__ __forceinline__ int swzX(int c, int t) {  // B fragment: X[c][t]
  int r = c & 31;
  int g = (r >> 3) & 1;
  int j = (r & 7) | ((r >> 4) << 3);
  int L = (g << 4) | (t & 15);
  return (((c >> 5) << 2) + (t >> 4)) * 512 + L * 16 + j;
}

__device__ __forceinline__ void async_ld_b32(uint32_t lds_off, const float* base,
                                             uint32_t byte_off) {
  asm volatile("global_load_async_to_lds_b32 %0, %1, %2"
               :: "v"(lds_off), "v"(byte_off), "s"(base) : "memory");
}

// one wave: 16-row strip (all 128 K) x 64 tokens => 4 col-tiles x 4 k-steps of wmma
__device__ __forceinline__ void gemm_strip(const bf16_t* Wl, const bf16_t* Xl,
                                           v8f acc[4], int wave, int lane) {
#pragma unroll
  for (int ks = 0; ks < 4; ++ks) {
    v16bf a = *(const v16bf*)(Wl + (wave * 4 + ks) * 512 + lane * 16);
#pragma unroll
    for (int tc = 0; tc < 4; ++tc) {
      v16bf bm = *(const v16bf*)(Xl + (ks * 4 + tc) * 512 + lane * 16);
      acc[tc] = __builtin_amdgcn_wmma_f32_16x16x32_bf16(
          false, a, false, bm, (short)0, acc[tc], false, false);
    }
  }
}

// C/D tile layout: lane L -> col t = L&15; VGPR r -> row m = r + (L>>4)*8
__device__ __forceinline__ void store_qkv(v8f acc[4], bf16_t* dst,
                                          const float* bias, int wave, int lane) {
  int tl = lane & 15, gr = lane >> 4;
#pragma unroll
  for (int tc = 0; tc < 4; ++tc) {
    int t = tc * 16 + tl;
#pragma unroll
    for (int r = 0; r < 8; ++r) {
      int m = wave * 16 + gr * 8 + r;
      dst[t * 128 + m] = (bf16_t)(acc[tc][r] + bias[m]);
    }
  }
}

extern "C" __global__ void convert_weights(const float* __restrict__ inw,
                                           const float* __restrict__ outw,
                                           bf16_t* __restrict__ dst) {
  int i = blockIdx.x * blockDim.x + threadIdx.x;   // 65536 threads
  if (i < 49152) dst[i] = (bf16_t)inw[i];          // Wq|Wk|Wv
  int j = i - 49152;
  if (j >= 0 && j < 16384) dst[49152 + j] = (bf16_t)outw[j];  // Wout
}

extern "C" __global__ __launch_bounds__(256)
void cfa_fused(const float* __restrict__ f_t, const float* __restrict__ f_p,
               const float* __restrict__ f_n, const bf16_t* __restrict__ wbf,
               const float* __restrict__ in_b, const float* __restrict__ ob,
               const float* __restrict__ gamma, const float* __restrict__ beta,
               float* __restrict__ out) {
  extern __shared__ char smem[];
  bf16_t* sW    = (bf16_t*)(smem);
  bf16_t* sX    = (bf16_t*)(smem + 32768);
  float*  sXt   = (float*)(smem + 81920);
  bf16_t* sQ    = (bf16_t*)(smem + 114688);
  bf16_t* sK    = (bf16_t*)(smem + 131072);
  bf16_t* sV    = (bf16_t*)(smem + 180224);
  float*  sXln  = (float*)(smem + 180224);  // recycles sV after phase 2
  bf16_t* sCtx  = (bf16_t*)(smem + 229376);
  float*  sMu   = (float*)(smem + 245760);
  float*  sRs   = (float*)(smem + 246016);
  float*  sBias = (float*)(smem + 246272);  // in_b|out_b|gamma|beta

  const int tid = threadIdx.x;
  const int wave = tid >> 5, lane = tid & 31;
  const int tile = blockIdx.x;
  const int b  = tile >> 8;            // HW/NT = 256 tiles per batch image
  const int rt = tile & 255;
  const int h  = rt >> 1;              // 2 tiles per W-row
  const int w0 = (rt & 1) * NT;
  const long tbase = (long)b * C_DIM * HWSZ + h * 128 + w0;  // (c=0,t=0)

  // ---- residual f_t tile fp32 -> LDS via Tensor Data Mover ----
  // 2D tile: tile_dim0 = 64 contiguous fp32 (t), tile_dim1 = 128 rows (c),
  // row stride = H*W elements; TDM packs rows contiguously => sXt[c*64+t].
  if (wave == 0) {
    uint64_t ga = (uint64_t)(uintptr_t)(f_t + tbase);
    uint32_t la = (uint32_t)(uintptr_t)sXt;
    v4u g0;
    g0.x = 1u;                                   // count=1, is_restore=0
    g0.y = la;                                   // lds_addr [63:32]
    g0.z = (uint32_t)ga;                         // global_addr [95:64]
    g0.w = ((uint32_t)(ga >> 32) & 0x01FFFFFFu)  // global_addr [120:96]
           | (2u << 30);                         // type=2 (image) [127:126]
    v8i_t g1;
    g1[0] = (int)(2u << 16);     // workgroup_mask=0, data_size=2 (4 bytes)
    g1[1] = (int)(64u << 16);    // tensor_dim0 = 64        [63:48]
    g1[2] = (int)(128u << 16);   // tensor_dim1 = 128       [95:80]
    g1[3] = (int)(64u << 16);    // tile_dim0   = 64        [127:112]
    g1[4] = 128;                 // tile_dim1   = 128       [143:128]
    g1[5] = HWSZ;                // tensor_dim0_stride lo32 [191:160]
    g1[6] = 0;                   // stride hi / dim1_stride lo
    g1[7] = 0;                   // dim1_stride hi (unused for 2D)
    v4i_t gz4 = {0, 0, 0, 0};
    v8i_t gz8 = {0, 0, 0, 0, 0, 0, 0, 0};
    __builtin_amdgcn_tensor_load_to_lds(g0, g1, gz4, gz4, gz8, 0);
    __builtin_amdgcn_s_wait_tensorcnt(0);
  }

  // ---- per-channel vectors -> LDS via async DMA (ASYNCcnt path) ----
  {
    uint32_t bb = (uint32_t)(uintptr_t)sBias;
    for (int i = tid; i < 384; i += 256) async_ld_b32(bb + i * 4u, in_b, i * 4u);
    if (tid < 128) {
      async_ld_b32(bb + (384 + tid) * 4u, ob, tid * 4u);
      async_ld_b32(bb + (512 + tid) * 4u, gamma, tid * 4u);
      async_ld_b32(bb + (640 + tid) * 4u, beta, tid * 4u);
    }
  }

  // ---- stage the three frame tiles as bf16, WMMA-B swizzled ----
#pragma unroll
  for (int f = 0; f < 3; ++f) {
    const float* src = ((f == 0) ? f_t : (f == 1) ? f_p : f_n) + tbase;
    bf16_t* dst = sX + f * 8192;
    for (int i = tid; i < C_DIM * NT; i += 256) {
      int c = i >> 6, t = i & 63;
      dst[swzX(c, t)] = (bf16_t)src[c * HWSZ + t];
    }
  }
  // stage Wq (and prefetch Wk)
  for (int i = tid; i < 16384; i += 256) {
    __builtin_prefetch(wbf + 16384 + i, 0, 1);
    int m = i >> 7, c = i & 127;
    sW[swzW(m, c)] = wbf[i];
  }
  asm volatile("s_wait_asynccnt 0" ::: "memory");
  __syncthreads();

  const float* lb_in = sBias;          // in_proj_b (384)
  const float* lb_ob = sBias + 384;    // out_b
  const float* lb_g  = sBias + 512;    // gamma
  const float* lb_b  = sBias + 640;    // beta

  const v8f vzero = {0.f, 0.f, 0.f, 0.f, 0.f, 0.f, 0.f, 0.f};
  v8f acc[4];

  // ---- Q = Wq @ Xt^T ----
  acc[0] = vzero; acc[1] = vzero; acc[2] = vzero; acc[3] = vzero;
  gemm_strip(sW, sX, acc, wave, lane);
  store_qkv(acc, sQ, lb_in, wave, lane);
  __syncthreads();

  // ---- K_s = Wk @ Xs^T ----
  for (int i = tid; i < 16384; i += 256) {
    __builtin_prefetch(wbf + 32768 + i, 0, 1);
    int m = i >> 7, c = i & 127;
    sW[swzW(m, c)] = wbf[16384 + i];
  }
  __syncthreads();
#pragma unroll
  for (int f = 0; f < 3; ++f) {
    acc[0] = vzero; acc[1] = vzero; acc[2] = vzero; acc[3] = vzero;
    gemm_strip(sW, sX + f * 8192, acc, wave, lane);
    store_qkv(acc, sK + f * 8192, lb_in + 128, wave, lane);
  }
  __syncthreads();

  // ---- V_s = Wv @ Xs^T ----
  for (int i = tid; i < 16384; i += 256) {
    __builtin_prefetch(wbf + 49152 + i, 0, 1);
    int m = i >> 7, c = i & 127;
    sW[swzW(m, c)] = wbf[32768 + i];
  }
  __syncthreads();
#pragma unroll
  for (int f = 0; f < 3; ++f) {
    acc[0] = vzero; acc[1] = vzero; acc[2] = vzero; acc[3] = vzero;
    gemm_strip(sW, sX + f * 8192, acc, wave, lane);
    store_qkv(acc, sV + f * 8192, lb_in + 256, wave, lane);
  }
  __syncthreads();

  // ---- stage Wout (sW free now), then per-(token,head) softmax over 3 frames ----
  for (int i = tid; i < 16384; i += 256) {
    int m = i >> 7, c = i & 127;
    sW[swzW(m, c)] = wbf[49152 + i];
  }
  {
    int t = tid >> 2, hh = tid & 3;
    const bf16_t* qp = sQ + t * 128 + hh * 32;
    float q[32];
#pragma unroll
    for (int d = 0; d < 32; ++d) q[d] = (float)qp[d];
    float sc[3];
#pragma unroll
    for (int s = 0; s < 3; ++s) {
      const bf16_t* kp = sK + s * 8192 + t * 128 + hh * 32;
      float dd = 0.f;
#pragma unroll
      for (int d = 0; d < 32; ++d) dd += q[d] * (float)kp[d];
      sc[s] = dd * 0.17677669529663689f;  // 1/sqrt(32)
    }
    float mx = fmaxf(sc[0], fmaxf(sc[1], sc[2]));
    float e0 = __expf(sc[0] - mx), e1 = __expf(sc[1] - mx), e2 = __expf(sc[2] - mx);
    float inv = 1.f / (e0 + e1 + e2);
    e0 *= inv; e1 *= inv; e2 *= inv;
    const bf16_t* v0 = sV + 0 * 8192 + t * 128 + hh * 32;
    const bf16_t* v1 = sV + 1 * 8192 + t * 128 + hh * 32;
    const bf16_t* v2 = sV + 2 * 8192 + t * 128 + hh * 32;
#pragma unroll
    for (int d = 0; d < 32; ++d) {
      float cv = e0 * (float)v0[d] + e1 * (float)v1[d] + e2 * (float)v2[d];
      sCtx[swzX(hh * 32 + d, t)] = (bf16_t)cv;
    }
  }
  __syncthreads();

  // ---- out = Wout @ ctx^T + out_b + residual (TDM-loaded fp32) ----
  acc[0] = vzero; acc[1] = vzero; acc[2] = vzero; acc[3] = vzero;
  gemm_strip(sW, sCtx, acc, wave, lane);
  {
    int tl = lane & 15, gr = lane >> 4;
#pragma unroll
    for (int tc = 0; tc < 4; ++tc) {
      int t = tc * 16 + tl;
#pragma unroll
      for (int r = 0; r < 8; ++r) {
        int m = wave * 16 + gr * 8 + r;
        sXln[m * 64 + t] = acc[tc][r] + lb_ob[m] + sXt[m * 64 + t];
      }
    }
  }
  __syncthreads();

  // ---- LayerNorm stats (bank-conflict-free: fixed t => fixed bank per lane) ----
  if (tid < 64) {
    float s1 = 0.f;
    for (int c = 0; c < 128; ++c) s1 += sXln[c * 64 + tid];
    float mu = s1 * (1.f / 128.f);
    float s2 = 0.f;
    for (int c = 0; c < 128; ++c) {
      float d = sXln[c * 64 + tid] - mu;
      s2 += d * d;
    }
    sMu[tid] = mu;
    sRs[tid] = rsqrtf(s2 * (1.f / 128.f) + 1e-5f);
  }
  __syncthreads();

  // ---- normalize + coalesced channel-major store ----
  for (int i = tid; i < C_DIM * NT; i += 256) {
    int c = i >> 6, t = i & 63;
    float v = (sXln[c * 64 + t] - sMu[t]) * sRs[t] * lb_g[c] + lb_b[c];
    out[tbase + c * HWSZ + t] = v;
  }
}

extern "C" void kernel_launch(void* const* d_in, const int* in_sizes, int n_in,
                              void* d_out, int out_size, void* d_ws, size_t ws_size,
                              hipStream_t stream) {
  const float* f_t  = (const float*)d_in[0];
  const float* f_p  = (const float*)d_in[1];
  const float* f_n  = (const float*)d_in[2];
  const float* in_w = (const float*)d_in[3];
  const float* in_b = (const float*)d_in[4];
  const float* ow   = (const float*)d_in[5];
  const float* obv  = (const float*)d_in[6];
  const float* gmm  = (const float*)d_in[7];
  const float* bta  = (const float*)d_in[8];
  bf16_t* wbf = (bf16_t*)d_ws;  // 65536 bf16 = 128 KB scratch

  convert_weights<<<256, 256, 0, stream>>>(in_w, ow, wbf);
  cfa_fused<<<2048, 256, SMEM_BYTES, stream>>>(f_t, f_p, f_n, wbf, in_b, obv,
                                               gmm, bta, (float*)d_out);
}
